// SwinTransformer3DBlock_52750788329878
// MI455X (gfx1250) — matrix operand order repaired and checked by reading, not compiled, not run on hardware
//
#include <hip/hip_runtime.h>
#include <hip/hip_bf16.h>

typedef __attribute__((ext_vector_type(16))) _Float16 v16h;
typedef __attribute__((ext_vector_type(8)))  float    v8f;

#define WMMA(a, b, c) __builtin_amdgcn_wmma_f32_16x16x32_f16( \
    false, (a), false, (b), (short)0, (c), false, false)

// ---------------------------------------------------------------------------
// Fragment loaders (contiguous k order for A/B, wave32)
// A (16x32 f16, row-major src, row stride ld):
//   elem i of lane L -> row = L%16, col = (L/16)*16 + i  (16 contiguous halves)
// B (32x16 f16) built from row-major W(N,K): B[k][n] = W[n][k]:
//   elem i of lane L -> n = L%16, k = (L/16)*16 + i  (16 contiguous halves)
// C/D (16x16 f32): vgpr r of lane L -> M = r + (L/16)*8, N = L%16
// ---------------------------------------------------------------------------
__device__ __forceinline__ v16h load_a_frag(const _Float16* base, int ld, int lane) {
    const _Float16* p = base + (size_t)(lane & 15) * ld + ((lane >> 4) << 4);
    v16h a;
#pragma unroll
    for (int i = 0; i < 8; ++i) a[i] = p[i];
#pragma unroll
    for (int i = 0; i < 8; ++i) a[i + 8] = p[8 + i];
    return a;
}

__device__ __forceinline__ v16h load_bt_frag(const _Float16* base, int ld, int lane) {
    const _Float16* p = base + (size_t)(lane & 15) * ld + ((lane >> 4) << 4);
    v16h b;
#pragma unroll
    for (int i = 0; i < 16; ++i) b[i] = p[i];
    return b;
}

__device__ __forceinline__ float wave_sum(float v) {
#pragma unroll
    for (int o = 16; o > 0; o >>= 1) v += __shfl_xor(v, o, 32);
    return v;
}

__device__ __forceinline__ float wave_max(float v) {
#pragma unroll
    for (int o = 16; o > 0; o >>= 1) v = fmaxf(v, __shfl_xor(v, o, 32));
    return v;
}

// ---------------------------------------------------------------------------
// K0: f32 -> f16 weight convert
// ---------------------------------------------------------------------------
__global__ void k_f2h(const float* __restrict__ s, _Float16* __restrict__ d, int n) {
    int i = blockIdx.x * blockDim.x + threadIdx.x;
    if (i < n) d[i] = (_Float16)s[i];
}

// K0b: zero (uint4 granularity)
__global__ void k_zero16(uint4* __restrict__ d, int n4) {
    int i = blockIdx.x * blockDim.x + threadIdx.x;
    if (i < n4) d[i] = make_uint4(0u, 0u, 0u, 0u);
}

// ---------------------------------------------------------------------------
// K1: LN1 + cyclic shift (-1,-3,-3) + window partition -> xw f16 (2048,112,128)
// one wave per padded-window row
// ---------------------------------------------------------------------------
__global__ void k_ln1_window(const float* __restrict__ x, const float* __restrict__ g,
                             const float* __restrict__ be, _Float16* __restrict__ xw) {
    int gid  = blockIdx.x * blockDim.x + threadIdx.x;
    int row  = gid >> 5;
    int lane = gid & 31;
    if (row >= 2048 * 112) return;
    _Float16* dst = xw + (size_t)row * 128 + lane * 4;
    int nn = row % 112;
    if (nn >= 98) {
#pragma unroll
        for (int i = 0; i < 4; ++i) dst[i] = (_Float16)0.f;
        return;
    }
    int win = row / 112;
    int b = win >> 8, wib = win & 255;
    int wt = wib >> 6, wh = (wib >> 3) & 7, ww = wib & 7;
    int nt = nn / 49, rr = nn % 49, nh = rr / 7, nw = rr % 7;
    int t = (wt * 2 + nt + 1) & 7;
    int h = wh * 7 + nh + 3; if (h >= 56) h -= 56;
    int w = ww * 7 + nw + 3; if (w >= 56) w -= 56;
    const float* src = x + ((((size_t)b * 8 + t) * 56 + h) * 56 + w) * 128 + lane * 4;
    float4 v = *(const float4*)src;
    float mean = wave_sum(v.x + v.y + v.z + v.w) * 0.0078125f;
    float d0 = v.x - mean, d1 = v.y - mean, d2 = v.z - mean, d3 = v.w - mean;
    float var = wave_sum(d0 * d0 + d1 * d1 + d2 * d2 + d3 * d3) * 0.0078125f;
    float rs = rsqrtf(var + 1e-5f);
    int c = lane * 4;
    dst[0] = (_Float16)(d0 * rs * g[c + 0] + be[c + 0]);
    dst[1] = (_Float16)(d1 * rs * g[c + 1] + be[c + 1]);
    dst[2] = (_Float16)(d2 * rs * g[c + 2] + be[c + 2]);
    dst[3] = (_Float16)(d3 * rs * g[c + 3] + be[c + 3]);
}

// ---------------------------------------------------------------------------
// K2: QKV GEMM (M=229376, N=384, K=128), wave tile 16x64 -> scatter q,k,vT
// ---------------------------------------------------------------------------
__global__ void k_qkv(const _Float16* __restrict__ xw, const _Float16* __restrict__ wq,
                      const float* __restrict__ qb, _Float16* __restrict__ qo,
                      _Float16* __restrict__ ko, _Float16* __restrict__ vt) {
    int wave = threadIdx.x >> 5, lane = threadIdx.x & 31;
    int wi = blockIdx.x * 8 + wave;
    if (wi >= 14336 * 6) return;
    int mt = wi / 6, nq = wi % 6;
    const _Float16* A = xw + (size_t)mt * 16 * 128;
    v8f c0 = {}, c1 = {}, c2 = {}, c3 = {};
#pragma unroll
    for (int kk = 0; kk < 4; ++kk) {
        v16h a  = load_a_frag(A + kk * 32, 128, lane);
        const _Float16* wb = wq + (size_t)(nq * 64) * 128 + kk * 32;
        __builtin_prefetch(wb + 64 * 128, 0, 1);
        v16h b0 = load_bt_frag(wb +  0 * 128, 128, lane);
        v16h b1 = load_bt_frag(wb + 16 * 128, 128, lane);
        v16h b2 = load_bt_frag(wb + 32 * 128, 128, lane);
        v16h b3 = load_bt_frag(wb + 48 * 128, 128, lane);
        c0 = WMMA(a, b0, c0);
        c1 = WMMA(a, b1, c1);
        c2 = WMMA(a, b2, c2);
        c3 = WMMA(a, b3, c3);
    }
    int mbase = mt * 16 + ((lane >> 4) << 3);
    auto epi = [&](v8f c, int ncol) {
        float bias = qb[ncol];
        int s = ncol >> 7, rem = ncol & 127, hd = rem >> 5, dd = rem & 31;
#pragma unroll
        for (int r = 0; r < 8; ++r) {
            int m = mbase + r;
            int win = m / 112, nn = m % 112;
            float v = c[r] + bias;
            if (s == 0) {
                qo[(((size_t)win * 4 + hd) * 112 + nn) * 32 + dd] =
                    (_Float16)(v * 0.17677669529663687f);
            } else if (s == 1) {
                ko[(((size_t)win * 4 + hd) * 112 + nn) * 32 + dd] = (_Float16)v;
            } else {
                vt[(((size_t)win * 4 + hd) * 32 + dd) * 128 + nn] = (_Float16)v;
            }
        }
    };
    int c0n = nq * 64 + (lane & 15);
    epi(c0, c0n); epi(c1, c0n + 16); epi(c2, c0n + 32); epi(c3, c0n + 48);
}

// ---------------------------------------------------------------------------
// K3: attention per (window, head). N=98 padded to 112 (S) / 128 (P K-dim)
// ---------------------------------------------------------------------------
__global__ void k_attn(const _Float16* __restrict__ q, const _Float16* __restrict__ kx,
                       const _Float16* __restrict__ vt, const float* __restrict__ rpb,
                       const float* __restrict__ mask, _Float16* __restrict__ ao) {
    __shared__ __align__(16) float    S[16 * 112];
    __shared__ __align__(16) _Float16 P[16 * 128];
    int wave = threadIdx.x >> 5, lane = threadIdx.x & 31;
    int win = blockIdx.x >> 2, h = blockIdx.x & 3;
    const _Float16* Qp = q  + ((size_t)win * 4 + h) * 112 * 32;
    const _Float16* Kp = kx + ((size_t)win * 4 + h) * 112 * 32;
    const _Float16* Vp = vt + ((size_t)win * 4 + h) * 32 * 128;
    const float* mrow = mask + (size_t)(win & 255) * 98 * 98;
    for (int mt = 0; mt < 7; ++mt) {
        if (wave < 7) {                 // S = Q K^T (16x112 strip)
            v16h a = load_a_frag(Qp + mt * 16 * 32, 32, lane);
            v16h b = load_bt_frag(Kp + wave * 16 * 32, 32, lane);
            v8f c = {};
            c = WMMA(a, b, c);
            int lr0 = (lane >> 4) << 3;
            int j = wave * 16 + (lane & 15);
#pragma unroll
            for (int r = 0; r < 8; ++r) {
                int i = mt * 16 + lr0 + r;
                float sv = -1e30f;
                if (i < 98 && j < 98) {
                    int ti = i / 49, ri = i % 49, hi = ri / 7, wi = ri % 7;
                    int tj = j / 49, rj = j % 49, hj = rj / 7, wj = rj % 7;
                    int idx = (ti - tj + 1) * 169 + (hi - hj + 6) * 13 + (wi - wj + 6);
                    sv = c[r] + rpb[idx * 4 + h] + mrow[i * 98 + j];
                }
                S[(lr0 + r) * 112 + j] = sv;
            }
        }
        __syncthreads();
        {   // row softmax -> P(f16, cols padded to 128); wave owns 2 rows,
            // lanes 0..27 own 4 cols each; wave32 shuffle reductions
            int lr = wave * 2;
#pragma unroll
            for (int rr = 0; rr < 2; ++rr, ++lr) {
                int i = mt * 16 + lr;
                int c0 = lane * 4;
                bool act = (lane < 28);
                float e0 = act ? S[lr * 112 + c0 + 0] : -3e30f;
                float e1 = act ? S[lr * 112 + c0 + 1] : -3e30f;
                float e2 = act ? S[lr * 112 + c0 + 2] : -3e30f;
                float e3 = act ? S[lr * 112 + c0 + 3] : -3e30f;
                float mx = wave_max(fmaxf(fmaxf(e0, e1), fmaxf(e2, e3)));
                e0 = __expf(e0 - mx); e1 = __expf(e1 - mx);
                e2 = __expf(e2 - mx); e3 = __expf(e3 - mx);
                float sum = wave_sum(act ? (e0 + e1 + e2 + e3) : 0.f);
                float inv = (i < 98) ? (1.f / sum) : 0.f;   // padded rows -> P = 0
                if (act) {
                    P[lr * 128 + c0 + 0] = (_Float16)(e0 * inv);
                    P[lr * 128 + c0 + 1] = (_Float16)(e1 * inv);
                    P[lr * 128 + c0 + 2] = (_Float16)(e2 * inv);
                    P[lr * 128 + c0 + 3] = (_Float16)(e3 * inv);
                } else {                 // lanes 28..31 zero P cols 112..127
                    P[lr * 128 + c0 + 0] = (_Float16)0.f;
                    P[lr * 128 + c0 + 1] = (_Float16)0.f;
                    P[lr * 128 + c0 + 2] = (_Float16)0.f;
                    P[lr * 128 + c0 + 3] = (_Float16)0.f;
                }
            }
        }
        __syncthreads();
        if (wave < 2) {                 // O = P @ V (16x32 strip), K = 128
            v8f o = {};
#pragma unroll
            for (int kk = 0; kk < 4; ++kk) {
                v16h a = load_a_frag(P + kk * 32, 128, lane);
                v16h b = load_bt_frag(Vp + wave * 16 * 128 + kk * 32, 128, lane);
                o = WMMA(a, b, o);
            }
            int lr0 = (lane >> 4) << 3;
            int dd = wave * 16 + (lane & 15);
#pragma unroll
            for (int r = 0; r < 8; ++r) {
                int m = mt * 16 + lr0 + r;
                ao[((size_t)win * 112 + m) * 128 + h * 32 + dd] = (_Float16)o[r];
            }
        }
        __syncthreads();
    }
}

// ---------------------------------------------------------------------------
// K4: proj GEMM + window reverse + roll-back + residual -> y (f32)
// ---------------------------------------------------------------------------
__global__ void k_proj(const _Float16* __restrict__ ao, const _Float16* __restrict__ wp,
                       const float* __restrict__ pb, const float* __restrict__ x,
                       float* __restrict__ y) {
    int wave = threadIdx.x >> 5, lane = threadIdx.x & 31;
    int wi = blockIdx.x * 8 + wave;
    if (wi >= 14336 * 2) return;
    int mt = wi >> 1, nq = wi & 1;
    const _Float16* A = ao + (size_t)mt * 16 * 128;
    v8f c0 = {}, c1 = {}, c2 = {}, c3 = {};
#pragma unroll
    for (int kk = 0; kk < 4; ++kk) {
        v16h a = load_a_frag(A + kk * 32, 128, lane);
        const _Float16* wb = wp + (size_t)(nq * 64) * 128 + kk * 32;
        v16h b0 = load_bt_frag(wb +  0 * 128, 128, lane);
        v16h b1 = load_bt_frag(wb + 16 * 128, 128, lane);
        v16h b2 = load_bt_frag(wb + 32 * 128, 128, lane);
        v16h b3 = load_bt_frag(wb + 48 * 128, 128, lane);
        c0 = WMMA(a, b0, c0);
        c1 = WMMA(a, b1, c1);
        c2 = WMMA(a, b2, c2);
        c3 = WMMA(a, b3, c3);
    }
    int mbase = mt * 16 + ((lane >> 4) << 3);
    auto epi = [&](v8f c, int n) {
        float bb = pb[n];
#pragma unroll
        for (int r = 0; r < 8; ++r) {
            int m = mbase + r;
            int win = m / 112, nn = m % 112;
            if (nn < 98) {
                int b = win >> 8, wib = win & 255;
                int wt = wib >> 6, wh = (wib >> 3) & 7, ww = wib & 7;
                int nt = nn / 49, rr = nn % 49, nh = rr / 7, nw = rr % 7;
                int t = (wt * 2 + nt + 1) & 7;
                int hh = wh * 7 + nh + 3; if (hh >= 56) hh -= 56;
                int w2 = ww * 7 + nw + 3; if (w2 >= 56) w2 -= 56;
                size_t di = ((((size_t)b * 8 + t) * 56 + hh) * 56 + w2) * 128 + n;
                y[di] = x[di] + c[r] + bb;
            }
        }
    };
    int c0n = nq * 64 + (lane & 15);
    epi(c0, c0n); epi(c1, c0n + 16); epi(c2, c0n + 32); epi(c3, c0n + 48);
}

// ---------------------------------------------------------------------------
// K5: LN2 -> xn2 (f16), one wave per token
// ---------------------------------------------------------------------------
__global__ void k_ln2(const float* __restrict__ y, const float* __restrict__ g,
                      const float* __restrict__ be, _Float16* __restrict__ xn) {
    int gid = blockIdx.x * blockDim.x + threadIdx.x;
    int row = gid >> 5, lane = gid & 31;
    if (row >= 200704) return;
    const float* src = y + (size_t)row * 128 + lane * 4;
    float4 v = *(const float4*)src;
    float mean = wave_sum(v.x + v.y + v.z + v.w) * 0.0078125f;
    float d0 = v.x - mean, d1 = v.y - mean, d2 = v.z - mean, d3 = v.w - mean;
    float var = wave_sum(d0 * d0 + d1 * d1 + d2 * d2 + d3 * d3) * 0.0078125f;
    float rs = rsqrtf(var + 1e-5f);
    int c = lane * 4;
    _Float16* dst = xn + (size_t)row * 128 + lane * 4;
    dst[0] = (_Float16)(d0 * rs * g[c + 0] + be[c + 0]);
    dst[1] = (_Float16)(d1 * rs * g[c + 1] + be[c + 1]);
    dst[2] = (_Float16)(d2 * rs * g[c + 2] + be[c + 2]);
    dst[3] = (_Float16)(d3 * rs * g[c + 3] + be[c + 3]);
}

// ---------------------------------------------------------------------------
// K6: fused MLP: out = y + gelu(xn2 @ fc1^T + b1) @ fc2^T + b2
// 32-row strip per block; hid (512) processed in 8 chunks of 64 via LDS.
// XN strip staged with gfx1250 async global->LDS copies (ASYNCcnt path).
// ---------------------------------------------------------------------------
__global__ void k_mlp(const _Float16* __restrict__ xn, const _Float16* __restrict__ w1,
                      const float* __restrict__ b1, const _Float16* __restrict__ w2,
                      const float* __restrict__ b2, const float* __restrict__ y,
                      float* __restrict__ out) {
    __shared__ __align__(16) _Float16 XN[32 * 128];
    __shared__ __align__(16) _Float16 HS[32 * 72];
    int wave = threadIdx.x >> 5, lane = threadIdx.x & 31;
    size_t row0 = (size_t)blockIdx.x * 32;
    {   // stage XN strip into LDS: 2 async B128 copies per thread.
        // LDS byte address = low 32 bits of the flat shared-aperture address;
        // the instruction offset applies to both global and LDS sides.
        int t = threadIdx.x;
        int r = t >> 3, cb = (t & 7) * 16;
        const _Float16* gsrc = xn + (row0 + r) * 128 + cb;
        unsigned ldst = (unsigned)(uintptr_t)(&XN[r * 128 + cb]);
        asm volatile(
            "global_load_async_to_lds_b128 %0, %1, off\n\t"
            "global_load_async_to_lds_b128 %0, %1, off offset:16"
            :: "v"(ldst), "v"(gsrc) : "memory");
        asm volatile("s_wait_asynccnt 0x0" ::: "memory");
    }
    __syncthreads();
    int mh = wave & 1;       // m-tile (stage1 and stage2)
    int nt4 = wave >> 1;     // hid-tile (stage1) / 32-col group (stage2)
    v8f accA = {}, accB = {};
    for (int hc = 0; hc < 8; ++hc) {
        {   // stage1: H chunk = gelu(XN @ fc1^T) -> LDS
            v8f hA = {};
            const _Float16* Ab = XN + mh * 16 * 128;
            const _Float16* wb = w1 + (size_t)(hc * 64 + nt4 * 16) * 128;
            __builtin_prefetch(wb + 64 * 128, 0, 1);
#pragma unroll
            for (int kk = 0; kk < 4; ++kk) {
                v16h a = load_a_frag(Ab + kk * 32, 128, lane);
                v16h b = load_bt_frag(wb + kk * 32, 128, lane);
                hA = WMMA(a, b, hA);
            }
            int hcol = hc * 64 + nt4 * 16 + (lane & 15);
            float bb = b1[hcol];
            int lr = mh * 16 + ((lane >> 4) << 3);
#pragma unroll
            for (int r = 0; r < 8; ++r) {
                float v = hA[r] + bb;
                v = 0.5f * v * (1.f + erff(v * 0.70710678118654752f));
                HS[(lr + r) * 72 + nt4 * 16 + (lane & 15)] = (_Float16)v;
            }
        }
        __syncthreads();
        {   // stage2: accumulate H @ fc2^T
            const _Float16* Pb = HS + mh * 16 * 72;
#pragma unroll
            for (int kk = 0; kk < 2; ++kk) {
                v16h a  = load_a_frag(Pb + kk * 32, 72, lane);
                v16h bA = load_bt_frag(w2 + (size_t)(nt4 * 32     ) * 512 + hc * 64 + kk * 32, 512, lane);
                v16h bB = load_bt_frag(w2 + (size_t)(nt4 * 32 + 16) * 512 + hc * 64 + kk * 32, 512, lane);
                accA = WMMA(a, bA, accA);
                accB = WMMA(a, bB, accB);
            }
        }
        __syncthreads();
    }
    int lr = mh * 16 + ((lane >> 4) << 3);
    int nA = nt4 * 32 + (lane & 15), nB = nA + 16;
    float bbA = b2[nA], bbB = b2[nB];
#pragma unroll
    for (int r = 0; r < 8; ++r) {
        size_t m = row0 + lr + r;
        out[m * 128 + nA] = y[m * 128 + nA] + accA[r] + bbA;
        out[m * 128 + nB] = y[m * 128 + nB] + accB[r] + bbB;
    }
}

// ---------------------------------------------------------------------------
extern "C" void kernel_launch(void* const* d_in, const int* in_sizes, int n_in,
                              void* d_out, int out_size, void* d_ws, size_t ws_size,
                              hipStream_t stream) {
    const float* x     = (const float*)d_in[0];
    const float* maskm = (const float*)d_in[1];
    const float* ln1g  = (const float*)d_in[2];
    const float* ln1b  = (const float*)d_in[3];
    const float* qkvw  = (const float*)d_in[4];
    const float* qkvb  = (const float*)d_in[5];
    const float* rpb   = (const float*)d_in[6];
    const float* projw = (const float*)d_in[7];
    const float* projb = (const float*)d_in[8];
    const float* ln2g  = (const float*)d_in[9];
    const float* ln2b  = (const float*)d_in[10];
    const float* fc1w  = (const float*)d_in[11];
    const float* fc1b  = (const float*)d_in[12];
    const float* fc2w  = (const float*)d_in[13];
    const float* fc2b  = (const float*)d_in[14];
    float* out = (float*)d_out;

    char* ws = (char*)d_ws;
    _Float16* w_qkv  = (_Float16*)(ws + 0);              // 384*128
    _Float16* w_proj = (_Float16*)(ws + 98304);          // 128*128
    _Float16* w_fc1  = (_Float16*)(ws + 131072);         // 512*128
    _Float16* w_fc2  = (_Float16*)(ws + 262144);         // 128*512
    _Float16* xw     = (_Float16*)(ws + 393216);         // 2048*112*128 (reused as attn_out)
    _Float16* qbuf   = (_Float16*)(ws + 59113472ull);    // 2048*4*112*32 (reused as xn2)
    _Float16* kbuf   = (_Float16*)(ws + 117833728ull);   // 2048*4*112*32
    _Float16* vtbuf  = (_Float16*)(ws + 176553984ull);   // 2048*4*32*128
    float*    ybuf   = (float*)   (ws + 243662848ull);   // 200704*128

    // K0: weights -> f16 ; zero vT (padding cols must be finite)
    k_f2h<<<(49152 + 255) / 256, 256, 0, stream>>>(qkvw,  w_qkv,  49152);
    k_f2h<<<(16384 + 255) / 256, 256, 0, stream>>>(projw, w_proj, 16384);
    k_f2h<<<(65536 + 255) / 256, 256, 0, stream>>>(fc1w,  w_fc1,  65536);
    k_f2h<<<(65536 + 255) / 256, 256, 0, stream>>>(fc2w,  w_fc2,  65536);
    k_zero16<<<16384, 256, 0, stream>>>((uint4*)vtbuf, 4194304);

    // K1: LN1 + shift + window partition
    k_ln1_window<<<28672, 256, 0, stream>>>(x, ln1g, ln1b, xw);

    // K2: QKV GEMM
    k_qkv<<<10752, 256, 0, stream>>>(xw, w_qkv, qkvb, qbuf, kbuf, vtbuf);

    // K3: windowed attention (attn_out overwrites xw)
    k_attn<<<8192, 256, 0, stream>>>(qbuf, kbuf, vtbuf, rpb, maskm, xw);

    // K4: proj + window reverse + residual -> y
    k_proj<<<3584, 256, 0, stream>>>(xw, w_proj, projb, x, ybuf);

    // K5: LN2 -> xn2 (overwrites qbuf region)
    k_ln2<<<25088, 256, 0, stream>>>(ybuf, ln2g, ln2b, qbuf);

    // K6: fused MLP + residual -> out
    k_mlp<<<6272, 256, 0, stream>>>(qbuf, w_fc1, fc1b, w_fc2, fc2b, ybuf, out);
}
